// JKSAGE_90366111908392
// MI455X (gfx1250) — compile-verified, matrix operand order checked by reading, not run on hardware
//
#include <hip/hip_runtime.h>
#include <hip/hip_bf16.h>

// ---------------------------------------------------------------------------
// JK-SAGE on MI455X (gfx1250, wave32).
//  - scatter/aggregate: memory-bound f32 atomics (feature table is L2-resident)
//  - GEMMs: v_wmma_f32_16x16x32_bf16, f32 accumulation
//  - weights pre-packed into WMMA B-fragment layout (bf16) once per launch
//  - A-tile staging loads/conversions hoisted to kernel prologue
// ---------------------------------------------------------------------------

typedef __attribute__((ext_vector_type(16))) __bf16 v16bf;
typedef __attribute__((ext_vector_type(8)))  __bf16 v8bf;
typedef __attribute__((ext_vector_type(8)))  float  v8f;

#define D_HID 128

static __device__ __forceinline__ unsigned bfbits(float f) {
    // round-to-nearest-even f32 -> bf16 bits
    unsigned u = __builtin_bit_cast(unsigned, f);
    u += 0x7FFFu + ((u >> 16) & 1u);
    return u >> 16;
}
static __device__ __forceinline__ __bf16 f2bf(float f) {
    unsigned short h = (unsigned short)bfbits(f);
    return __builtin_bit_cast(__bf16, h);
}
static __device__ __forceinline__ unsigned pack_bf2(float lo, float hi) {
    return bfbits(lo) | (bfbits(hi) << 16);
}

// WMMA 16-bit A/B fragment K index for element i of lane (half = lane>>4):
// lanes 0-15 : elems 0..7 -> K 0..7,  elems 8..15 -> K 16..23
// lanes 16-31: elems 0..7 -> K 8..15, elems 8..15 -> K 24..31
static __device__ __forceinline__ int frag_k(int i, int half) {
    return ((i >> 3) << 4) + (half << 3) + (i & 7);
}

// ---------------------------------------------------------------------------
// Pack a SAGE layer's weights [wl;wr] (each 128x128 f32, row-major [k][n])
// into bf16 WMMA B fragments: frag f = (nt*8 + kk)*32 + lane, 16 elems each.
// kk 0..3 -> wl (kbase=(kk&3)*32), kk 4..7 -> wr. Total 8*8*32*16 = 32768.
// ---------------------------------------------------------------------------
__global__ void pack_layer_w(const float* __restrict__ wl,
                             const float* __restrict__ wr,
                             __bf16* __restrict__ out) {
    int tid = blockIdx.x * 256 + threadIdx.x;
    if (tid >= 8 * 8 * 32 * 16) return;
    int i    = tid & 15;
    int lane = (tid >> 4) & 31;
    int kk   = (tid >> 9) & 7;
    int nt   = tid >> 12;                 // 0..7
    int half = lane >> 4, row16 = lane & 15;
    int k = (kk & 3) * 32 + frag_k(i, half);
    int n = nt * 16 + row16;
    const float* W = (kk < 4) ? wl : wr;
    out[tid] = f2bf(W[k * D_HID + n]);
}

// ---------------------------------------------------------------------------
// Pack head weights wout [256 x 40] f32 into bf16 fragments, columns padded
// to 64 with zeros. frag f = (nt*8 + kk)*32 + lane; K row = kk*32 + frag_k.
// Total 4*8*32*16 = 16384.
// ---------------------------------------------------------------------------
__global__ void pack_out_w(const float* __restrict__ wout,
                           __bf16* __restrict__ out) {
    int tid = blockIdx.x * 256 + threadIdx.x;
    if (tid >= 4 * 8 * 32 * 16) return;
    int i    = tid & 15;
    int lane = (tid >> 4) & 31;
    int kk   = (tid >> 9) & 7;
    int nt   = tid >> 12;                 // 0..3
    int half = lane >> 4, row16 = lane & 15;
    int k = kk * 32 + frag_k(i, half);    // 0..255
    int n = nt * 16 + row16;              // 0..63
    out[tid] = (n < 40) ? f2bf(wout[k * 40 + n]) : f2bf(0.0f);
}

// ---------------------------------------------------------------------------
// Edge scatter: agg[dst] += x[src] (per 4-feature chunk), cnt[dst] += 1.
// ---------------------------------------------------------------------------
__global__ void sage_scatter(const int* __restrict__ src,
                             const int* __restrict__ dst,
                             const float* __restrict__ x,
                             float* __restrict__ agg,
                             float* __restrict__ cnt,
                             long long nedge) {
    long long tid = (long long)blockIdx.x * blockDim.x + threadIdx.x;
    long long e = tid >> 5;
    int ch = (int)(tid & 31);
    if (e >= nedge) return;
    int s = src[e];
    int d = dst[e];
    const float4 v = ((const float4*)x)[(size_t)s * 32 + ch];
    float* ap = agg + (size_t)d * D_HID + ch * 4;
    atomicAdd(ap + 0, v.x);
    atomicAdd(ap + 1, v.y);
    atomicAdd(ap + 2, v.z);
    atomicAdd(ap + 3, v.w);
    if (ch == 0) atomicAdd(&cnt[d], 1.0f);
}

// ---------------------------------------------------------------------------
// Fused SAGE layer: out = relu( (agg/max(cnt,1)) @ wl + xin @ wr + bias )
// Block = 256 threads (8 waves) -> 16 rows x 128 cols; wave w owns cols
// [16w,16w+16). K loop kk 0..7 over 32-deep steps (agg|wl then xin|wr).
// All A-staging global loads + bf16 packing hoisted to the prologue.
// ---------------------------------------------------------------------------
__global__ void sage_gemm(const float* __restrict__ agg,
                          const float* __restrict__ cnt,
                          const float* __restrict__ xin,
                          const v16bf* __restrict__ pw,
                          const float* __restrict__ bias,
                          float* __restrict__ out,
                          int nrows) {
    __shared__ __bf16 As[16 * 32];
    unsigned* As32 = (unsigned*)As;

    const int block_row = blockIdx.x * 16;
    const int lane  = threadIdx.x & 31;
    const int wave  = threadIdx.x >> 5;       // 0..7
    const int half  = lane >> 4;
    const int row16 = lane & 15;
    const int ncol  = wave * 16 + row16;

    // this thread stages the bf16 pair (sm, sk..sk+1) for every K step
    const int sm = threadIdx.x >> 4;          // 0..15
    const int sk = (threadIdx.x & 15) * 2;    // 0,2,..,30
    int smr = block_row + sm;
    if (smr > nrows - 1) smr = nrows - 1;
    const float inv = 1.0f / fmaxf(cnt[smr], 1.0f);

    // prologue: load + pack all 8 staging pairs (addresses fixed per thread)
    unsigned staged[8];
#pragma unroll
    for (int kk = 0; kk < 8; ++kk) {
        const float* srcp = (kk < 4) ? agg : xin;
        const float scale = (kk < 4) ? inv : 1.0f;
        float2 v = *(const float2*)(srcp + (size_t)smr * D_HID + (kk & 3) * 32 + sk);
        staged[kk] = pack_bf2(v.x * scale, v.y * scale);
    }

    v8f c = {};

#pragma unroll
    for (int kk = 0; kk < 8; ++kk) {
        __syncthreads();
        As32[threadIdx.x] = staged[kk];
        __syncthreads();

        // A fragment: two aligned 16B runs from LDS
        v8bf a0 = *(const v8bf*)(As + row16 * 32 + half * 8);
        v8bf a1 = *(const v8bf*)(As + row16 * 32 + 16 + half * 8);
        v16bf a = __builtin_shufflevector(a0, a1, 0, 1, 2, 3, 4, 5, 6, 7,
                                          8, 9, 10, 11, 12, 13, 14, 15);
        // B fragment: one contiguous 32B load from packed weights
        v16bf b = pw[(wave * 8 + kk) * 32 + lane];

        c = __builtin_amdgcn_wmma_f32_16x16x32_bf16(
                false, a, false, b, (short)0, c, false, false);
    }

    // epilogue: bias + ReLU. C/D: VGPR j -> M = half*8 + j, N = row16.
    const float bb = bias[ncol];
#pragma unroll
    for (int j = 0; j < 8; ++j) {
        int m = block_row + half * 8 + j;
        if (m < nrows) {
            float v = c[j] + bb;
            out[(size_t)m * D_HID + ncol] = v > 0.0f ? v : 0.0f;
        }
    }
}

// ---------------------------------------------------------------------------
// JK head: out[N,40] = h1 @ wout[0:128,:] + h2 @ wout[128:256,:] + bout
// Block = 128 threads (4 waves) -> 16 rows x 64 cols (cols >= 40 discarded;
// packed weights are zero there).
// ---------------------------------------------------------------------------
__global__ void jk_out_gemm(const float* __restrict__ h1,
                            const float* __restrict__ h2,
                            const v16bf* __restrict__ pwout,
                            const float* __restrict__ bout,
                            float* __restrict__ out,
                            int nrows) {
    __shared__ __bf16 As[16 * 32];
    unsigned* As32 = (unsigned*)As;

    const int NCLS = 40;
    const int block_row = blockIdx.x * 16;
    const int lane  = threadIdx.x & 31;
    const int wave  = threadIdx.x >> 5;       // 0..3
    const int half  = lane >> 4;
    const int row16 = lane & 15;
    const int ncol  = wave * 16 + row16;      // 0..63

    // stage two pairs per thread: rows sm0 and sm0+8
    const int sm0 = threadIdx.x >> 4;         // 0..7
    const int sk  = (threadIdx.x & 15) * 2;
    int smr0 = block_row + sm0;
    int smr1 = block_row + sm0 + 8;
    if (smr0 > nrows - 1) smr0 = nrows - 1;
    if (smr1 > nrows - 1) smr1 = nrows - 1;

    // prologue: load + pack all 16 staging pairs
    unsigned staged0[8], staged1[8];
#pragma unroll
    for (int kk = 0; kk < 8; ++kk) {
        const float* srcp = (kk < 4) ? h1 : h2;
        const int off = (kk & 3) * 32 + sk;
        float2 v0 = *(const float2*)(srcp + (size_t)smr0 * D_HID + off);
        float2 v1 = *(const float2*)(srcp + (size_t)smr1 * D_HID + off);
        staged0[kk] = pack_bf2(v0.x, v0.y);
        staged1[kk] = pack_bf2(v1.x, v1.y);
    }

    v8f c = {};

#pragma unroll
    for (int kk = 0; kk < 8; ++kk) {
        __syncthreads();
        As32[threadIdx.x]       = staged0[kk];
        As32[threadIdx.x + 128] = staged1[kk];
        __syncthreads();

        v8bf a0 = *(const v8bf*)(As + row16 * 32 + half * 8);
        v8bf a1 = *(const v8bf*)(As + row16 * 32 + 16 + half * 8);
        v16bf a = __builtin_shufflevector(a0, a1, 0, 1, 2, 3, 4, 5, 6, 7,
                                          8, 9, 10, 11, 12, 13, 14, 15);
        v16bf b = pwout[(wave * 8 + kk) * 32 + lane];

        c = __builtin_amdgcn_wmma_f32_16x16x32_bf16(
                false, a, false, b, (short)0, c, false, false);
    }

    if (ncol < NCLS) {
        const float bb = bout[ncol];
#pragma unroll
        for (int j = 0; j < 8; ++j) {
            int m = block_row + half * 8 + j;
            if (m < nrows) out[(size_t)m * NCLS + ncol] = c[j] + bb;
        }
    }
}

// ---------------------------------------------------------------------------
extern "C" void kernel_launch(void* const* d_in, const int* in_sizes, int n_in,
                              void* d_out, int out_size, void* d_ws, size_t ws_size,
                              hipStream_t stream) {
    const float* x    = (const float*)d_in[0];
    const int*   ei1  = (const int*)d_in[1];
    const int*   ei2  = (const int*)d_in[2];
    const float* wl1  = (const float*)d_in[3];
    const float* wr1  = (const float*)d_in[4];
    const float* b1   = (const float*)d_in[5];
    const float* wl2  = (const float*)d_in[6];
    const float* wr2  = (const float*)d_in[7];
    const float* b2   = (const float*)d_in[8];
    const float* wout = (const float*)d_in[9];
    const float* bout = (const float*)d_in[10];

    const int N = in_sizes[0] / D_HID;               // 100000
    const long long E = (long long)in_sizes[1] / 2;  // 1.6M

    // workspace layout
    float* agg  = (float*)d_ws;                               // N*128 f32
    float* h1   = agg  + (size_t)N * D_HID;                   // N*128 f32
    float* h2   = h1   + (size_t)N * D_HID;                   // N*128 f32
    float* cnt1 = h2   + (size_t)N * D_HID;                   // N f32
    float* cnt2 = cnt1 + (size_t)N;                           // N f32
    __bf16* pw1   = (__bf16*)(cnt2 + (size_t)N);              // 32768 bf16
    __bf16* pw2   = pw1 + 32768;                              // 32768 bf16
    __bf16* pwout = pw2 + 32768;                              // 16384 bf16

    const int gemm_blocks = (N + 15) / 16;
    const long long scat_threads = E * 32;
    const int scat_blocks = (int)((scat_threads + 255) / 256);

    // ---- pack weights into WMMA fragment layout ----
    pack_layer_w<<<128, 256, 0, stream>>>(wl1, wr1, pw1);
    pack_layer_w<<<128, 256, 0, stream>>>(wl2, wr2, pw2);
    pack_out_w<<<64, 256, 0, stream>>>(wout, pwout);

    // ---- layer 1 ----
    hipMemsetAsync(agg,  0, (size_t)N * D_HID * sizeof(float), stream);
    hipMemsetAsync(cnt1, 0, (size_t)N * sizeof(float), stream);
    sage_scatter<<<scat_blocks, 256, 0, stream>>>(ei1, ei1 + E, x, agg, cnt1, E);
    sage_gemm<<<gemm_blocks, 256, 0, stream>>>(agg, cnt1, x, (const v16bf*)pw1, b1, h1, N);

    // ---- layer 2 (reuse agg) ----
    hipMemsetAsync(agg,  0, (size_t)N * D_HID * sizeof(float), stream);
    hipMemsetAsync(cnt2, 0, (size_t)N * sizeof(float), stream);
    sage_scatter<<<scat_blocks, 256, 0, stream>>>(ei2, ei2 + E, h1, agg, cnt2, E);
    sage_gemm<<<gemm_blocks, 256, 0, stream>>>(agg, cnt2, h1, (const v16bf*)pw2, b2, h2, N);

    // ---- jumping-knowledge head ----
    jk_out_gemm<<<gemm_blocks, 128, 0, stream>>>(h1, h2, (const v16bf*)pwout, bout, (float*)d_out, N);
}
